// EncoderRNN_68796786147919
// MI455X (gfx1250) — matrix operand order repaired and checked by reading, not compiled, not run on hardware
//
#include <hip/hip_runtime.h>
#include <hip/hip_bf16.h>

// ---------------------------------------------------------------------------
// Bidirectional GRU encoder for MI455X (gfx1250), wave32 + WMMA f16->f32.
//   B=256, T=512, V=50000, E=H=256, DH=512, C=10
// Decomposition:
//   K1: convert 4 weight matrices [768,256] f32 -> f16 in ws
//   K2: grid (16 batch-tiles, 2 directions); each block scans all T steps.
//       Per step: gi = x @ Wih^T, gh = h @ Whh^T via v_wmma_f32_16x16x32_f16.
//       K-loop kept rolled (#pragma unroll 1) so B fragments stream from
//       L2 via global_load_b128 without register spills.
//   K3: classifier tanh(W1 rep + b1) -> softmax(W2 hid + b2)
// ---------------------------------------------------------------------------

typedef __attribute__((ext_vector_type(16))) _Float16 v16h;
typedef __attribute__((ext_vector_type(8)))  float    v8f;

#define B_   256
#define T_   512
#define E_   256
#define H_   256
#define H3_  768
#define DH_  512
#define C_   10
#define BM   16            // batch rows per workgroup
#define SR   264           // LDS row stride in halves (256 + 8 pad -> bank skew)
#define WMAT (H3_ * H_)    // 196608 elements per weight matrix

union Frag16 { uint4 q[2]; v16h v; _Float16 h[16]; };

// A fragment (16x32 f16, M=batch rows) from LDS.
// ISA layout: lane L holds row L%16; L<16 -> K in {0..7, 16..23},
//             L>=16 -> K in {8..15, 24..31}; VGPR v pairs K = 2v,2v+1 (+16 for v>=4).
__device__ __forceinline__ v16h load_afrag(const _Float16* S, int kbase, int lane) {
    int row = lane & 15;
    int ko  = (lane >> 4) << 3;                 // 0 or 8
    const _Float16* p = S + row * SR + kbase + ko;
    Frag16 f;
    f.q[0] = *(const uint4*)(p);                // K = kbase+ko    .. +7
    f.q[1] = *(const uint4*)(p + 16);           // K = kbase+ko+16 .. +23
    return f.v;
}

// B fragment (32x16 f16, N=weight rows == B^T columns) streamed from global (L2).
// ISA layout: lane L holds column L%16; L<16 -> K in {0..7, 16..23},
//             L>=16 -> K in {8..15, 24..31} (same order as A fragment).
__device__ __forceinline__ v16h load_bfrag(const _Float16* W, int wrow0, int kbase, int lane) {
    int col = lane & 15;
    int ko  = (lane >> 4) << 3;                 // 0 or 8
    const _Float16* p = W + (size_t)(wrow0 + col) * H_ + kbase + ko;
    Frag16 f;
    f.q[0] = *(const uint4*)(p);
    f.q[1] = *(const uint4*)(p + 16);
    return f.v;
}

__device__ __forceinline__ v8f wmma_f16(v16h a, v16h b, v8f c) {
    return __builtin_amdgcn_wmma_f32_16x16x32_f16(false, a, false, b, (short)0, c, false, false);
}

__device__ __forceinline__ float sigmoidf_(float x) { return 1.0f / (1.0f + __expf(-x)); }

// ---------------------------------------------------------------- K1: weights
__global__ __launch_bounds__(256)
void cvt_weights_kernel(const float* __restrict__ a, const float* __restrict__ b,
                        const float* __restrict__ c, const float* __restrict__ d,
                        _Float16* __restrict__ out) {
    int i = blockIdx.x * 256 + threadIdx.x;
    if (i >= 4 * WMAT) return;
    int m = i / WMAT, r = i - m * WMAT;
    const float* src = (m == 0) ? a : (m == 1) ? b : (m == 2) ? c : d;
    out[i] = (_Float16)src[r];
}

// ---------------------------------------------------------------- K2: GRU scan
__global__ __launch_bounds__(256)
void gru_scan_kernel(const int* __restrict__ seqs, const int* __restrict__ lens,
                     const float* __restrict__ emb,
                     const _Float16* __restrict__ WihF, const _Float16* __restrict__ WhhF,
                     const _Float16* __restrict__ WihB, const _Float16* __restrict__ WhhB,
                     const float* __restrict__ bih_f, const float* __restrict__ bhh_f,
                     const float* __restrict__ bih_b, const float* __restrict__ bhh_b,
                     float* __restrict__ hfOut, float* __restrict__ hbOut) {
    __shared__ _Float16 xS[BM * SR];
    __shared__ _Float16 hS[BM * SR];
    __shared__ float    hF[BM][H_];
    __shared__ float    bsumS[2 * H_];   // bih+bhh for r,z gates
    __shared__ float    binS[H_];        // bih for n gate
    __shared__ float    bhnS[H_];        // bhh for n gate
    __shared__ int      lenS[BM];

    const int tid  = threadIdx.x;
    const int w    = tid >> 5;           // wave id (wave32), 0..7
    const int lane = tid & 31;
    const int bx   = blockIdx.x;         // batch tile
    const int dir  = blockIdx.y;         // 0 = forward, 1 = backward

    const _Float16* Wih = dir ? WihB : WihF;
    const _Float16* Whh = dir ? WhhB : WhhF;
    const float*    bih = dir ? bih_b : bih_f;
    const float*    bhh = dir ? bhh_b : bhh_f;

    // ---- init shared state ----
    if (tid < BM) lenS[tid] = lens[bx * BM + tid];
    for (int i = tid; i < 2 * H_; i += 256) bsumS[i] = bih[i] + bhh[i];
    for (int i = tid; i < H_; i += 256) { binS[i] = bih[2 * H_ + i]; bhnS[i] = bhh[2 * H_ + i]; }
    for (int i = tid; i < BM * SR; i += 256) hS[i] = (_Float16)0.0f;
    {
        float* hf_ = &hF[0][0];
        for (int i = tid; i < BM * H_; i += 256) hf_[i] = 0.0f;
    }
    __syncthreads();

    const v8f zero8 = {0.f, 0.f, 0.f, 0.f, 0.f, 0.f, 0.f, 0.f};

    for (int t = 0; t < T_; ++t) {
        const int t_act = dir ? (T_ - 1 - t) : t;

        // ---- gather x tile (embedding + front-pad mask) into LDS as f16 ----
        {
            int row = tid >> 4, ch = tid & 15;         // 16 rows x 16 chunks of 16 cols
            int gb  = bx * BM + row;
            int len = lenS[row];
            _Float16* dst = xS + row * SR + ch * 16;
            Frag16 pk;
            if (t_act >= T_ - len) {
                int tok = seqs[gb * T_ + t_act];
                const float4* src = (const float4*)(emb + (size_t)tok * E_ + ch * 16);
#pragma unroll
                for (int q = 0; q < 4; ++q) {
                    float4 f = src[q];
                    pk.h[q * 4 + 0] = (_Float16)f.x; pk.h[q * 4 + 1] = (_Float16)f.y;
                    pk.h[q * 4 + 2] = (_Float16)f.z; pk.h[q * 4 + 3] = (_Float16)f.w;
                }
            } else {
                pk.q[0] = make_uint4(0u, 0u, 0u, 0u);
                pk.q[1] = make_uint4(0u, 0u, 0u, 0u);
            }
            *(uint4*)(dst)     = pk.q[0];              // two ds_store_b128
            *(uint4*)(dst + 8) = pk.q[1];
            // prefetch next step's embedding row (-> global_prefetch_b8)
            int tn = dir ? (t_act - 1) : (t_act + 1);
            if (tn >= 0 && tn < T_) {
                int tok2 = seqs[gb * T_ + tn];
                __builtin_prefetch(emb + (size_t)tok2 * E_ + ch * 16, 0, 1);
            }
        }
        __syncthreads();

        // ---- GEMMs: wave w owns h-columns [32w, 32w+32) -> 6 output tiles ----
        // acc[0..1]=r (gi+gh merged), acc[2..3]=z (merged), acc[4..5]=n_gi, acc[6..7]=n_gh
        v8f acc[8];
#pragma unroll
        for (int i = 0; i < 8; ++i) acc[i] = zero8;

        // CRITICAL: keep this loop rolled. Full unroll spills B fragments to
        // scratch (observed in round-1 asm); rolled, the live set fits in
        // <256 VGPRs and B frags stream from L2 via global_load_b128 clauses.
#pragma unroll 1
        for (int kk = 0; kk < 8; ++kk) {
            const int kb = kk * 32;
            v16h ax = load_afrag(xS, kb, lane);
            v16h ah = load_afrag(hS, kb, lane);
#pragma unroll
            for (int ct = 0; ct < 2; ++ct) {
                const int nt = (2 * w + ct) * 16;      // column base within H
                v16h bri = load_bfrag(Wih, nt, kb, lane);
                acc[0 + ct] = wmma_f16(ax, bri, acc[0 + ct]);
                v16h brh = load_bfrag(Whh, nt, kb, lane);
                acc[0 + ct] = wmma_f16(ah, brh, acc[0 + ct]);
                v16h bzi = load_bfrag(Wih, H_ + nt, kb, lane);
                acc[2 + ct] = wmma_f16(ax, bzi, acc[2 + ct]);
                v16h bzh = load_bfrag(Whh, H_ + nt, kb, lane);
                acc[2 + ct] = wmma_f16(ah, bzh, acc[2 + ct]);
                v16h bni = load_bfrag(Wih, 2 * H_ + nt, kb, lane);
                acc[4 + ct] = wmma_f16(ax, bni, acc[4 + ct]);
                v16h bnh = load_bfrag(Whh, 2 * H_ + nt, kb, lane);
                acc[6 + ct] = wmma_f16(ah, bnh, acc[6 + ct]);
            }
        }
        __syncthreads();   // all waves done reading hS/xS before h update

        // ---- gate nonlinearity + masked h update (register-local) ----
        // C/D layout: lane L, vgpr v -> M = v + 8*(L>=16), N = L%16
#pragma unroll
        for (int ct = 0; ct < 2; ++ct) {
            const int j   = (2 * w + ct) * 16 + (lane & 15);
            const float bsr = bsumS[j];
            const float bsz = bsumS[H_ + j];
            const float bin = binS[j];
            const float bhn = bhnS[j];
            const int   bl  = (lane >> 4) << 3;
#pragma unroll
            for (int v = 0; v < 8; ++v) {
                const int brow = bl + v;
                float r = sigmoidf_(acc[0 + ct][v] + bsr);
                float z = sigmoidf_(acc[2 + ct][v] + bsz);
                float n = tanhf(acc[4 + ct][v] + bin + r * (acc[6 + ct][v] + bhn));
                float hold = hF[brow][j];
                float hnew = (1.0f - z) * n + z * hold;
                float res  = (t_act < lenS[brow]) ? hnew : hold;   // blend mask
                hF[brow][j] = res;
                hS[brow * SR + j] = (_Float16)res;
            }
        }
        // next iteration's gather + __syncthreads orders hS writes before reads
    }
    __syncthreads();

    // ---- write final hidden state ----
    float* hOut = dir ? hbOut : hfOut;
    {
        int row = tid >> 4, ch = tid & 15;
#pragma unroll
        for (int q = 0; q < 16; ++q)
            hOut[(size_t)(bx * BM + row) * H_ + ch * 16 + q] = hF[row][ch * 16 + q];
    }
}

// ---------------------------------------------------------------- K3: classifier
__global__ __launch_bounds__(256)
void classifier_kernel(const float* __restrict__ hf, const float* __restrict__ hb,
                       const float* __restrict__ W1, const float* __restrict__ b1,
                       const float* __restrict__ W2, const float* __restrict__ b2,
                       float* __restrict__ out) {
    __shared__ float rep[2 * H_];
    __shared__ float hid[DH_];
    __shared__ float lg[C_];
    const int b = blockIdx.x, tid = threadIdx.x;

    rep[tid]       = hf[(size_t)b * H_ + tid];
    rep[H_ + tid]  = hb[(size_t)b * H_ + tid];
    __syncthreads();

    for (int d = tid; d < DH_; d += 256) {
        const float* wr = W1 + (size_t)d * (2 * H_);
        float acc = b1[d];
        for (int k = 0; k < 2 * H_; ++k) acc = __builtin_fmaf(wr[k], rep[k], acc);
        hid[d] = tanhf(acc);
    }
    __syncthreads();

    if (tid < C_) {
        const float* wr = W2 + (size_t)tid * DH_;
        float acc = b2[tid];
        for (int k = 0; k < DH_; ++k) acc = __builtin_fmaf(wr[k], hid[k], acc);
        lg[tid] = acc;
    }
    __syncthreads();

    if (tid == 0) {
        float mx = lg[0];
        for (int c = 1; c < C_; ++c) mx = fmaxf(mx, lg[c]);
        float e[C_], s = 0.0f;
        for (int c = 0; c < C_; ++c) { e[c] = __expf(lg[c] - mx); s += e[c]; }
        float inv = 1.0f / s;
        for (int c = 0; c < C_; ++c) out[(size_t)b * C_ + c] = e[c] * inv;
    }
}

// ---------------------------------------------------------------- launch
extern "C" void kernel_launch(void* const* d_in, const int* in_sizes, int n_in,
                              void* d_out, int out_size, void* d_ws, size_t ws_size,
                              hipStream_t stream) {
    (void)in_sizes; (void)n_in; (void)out_size; (void)ws_size;

    const int*   seqs  = (const int*)d_in[0];
    const int*   lens  = (const int*)d_in[1];
    const float* emb   = (const float*)d_in[2];
    const float* Wih_f = (const float*)d_in[3];
    const float* Whh_f = (const float*)d_in[4];
    const float* bih_f = (const float*)d_in[5];
    const float* bhh_f = (const float*)d_in[6];
    const float* Wih_b = (const float*)d_in[7];
    const float* Whh_b = (const float*)d_in[8];
    const float* bih_b = (const float*)d_in[9];
    const float* bhh_b = (const float*)d_in[10];
    const float* W1    = (const float*)d_in[11];
    const float* b1    = (const float*)d_in[12];
    const float* W2    = (const float*)d_in[13];
    const float* b2    = (const float*)d_in[14];

    // workspace layout
    _Float16* wH   = (_Float16*)d_ws;                       // 4 * WMAT halves
    _Float16* WihF = wH;
    _Float16* WhhF = wH + 1 * WMAT;
    _Float16* WihB = wH + 2 * WMAT;
    _Float16* WhhB = wH + 3 * WMAT;
    float* hf = (float*)((char*)d_ws + (size_t)4 * WMAT * sizeof(_Float16));
    float* hb = hf + (size_t)B_ * H_;

    // K1: f32 -> f16 weight conversion
    cvt_weights_kernel<<<(4 * WMAT + 255) / 256, 256, 0, stream>>>(
        Wih_f, Whh_f, Wih_b, Whh_b, wH);

    // K2: bidirectional GRU scan (16 batch tiles x 2 directions)
    gru_scan_kernel<<<dim3(16, 2), 256, 0, stream>>>(
        seqs, lens, emb, WihF, WhhF, WihB, WhhB,
        bih_f, bhh_f, bih_b, bhh_b, hf, hb);

    // K3: classifier + softmax
    classifier_kernel<<<B_, 256, 0, stream>>>(hf, hb, W1, b1, W2, b2, (float*)d_out);
}
